// MGCN_27968827032158
// MI455X (gfx1250) — matrix operand order repaired and checked by reading, not compiled, not run on hardware
//
#include <hip/hip_runtime.h>
#include <hip/hip_bf16.h>

// MGCN pipeline for MI455X (gfx1250, wave32, WMMA).
// All GEMMs go through v_wmma_f32_16x16x32_f16 (f16 operands, f32 accum),
// register-blocked 16x64 per wave, double-buffered LDS B tiles.
// Workspace requirement: ~26 MB.

#define Bd 8
#define Nd 512
#define Cd 256

typedef _Float16 v16h __attribute__((ext_vector_type(16)));
typedef _Float16 v8h  __attribute__((ext_vector_type(8)));
typedef float    v8f  __attribute__((ext_vector_type(8)));

enum { FLAG_RELU = 1, FLAG_F16OUT = 2, FLAG_BIAS = 4, FLAG_MASK = 8 };

// ---------------------------------------------------------------- converts
__global__ void f32_to_f16(const float* __restrict__ src, _Float16* __restrict__ dst, int n) {
    int i = blockIdx.x * blockDim.x + threadIdx.x;
    if (i < n) dst[i] = (_Float16)src[i];
}

// ew1 (2C x 32) -> packed B (C x 64) f16 : cols 0..31 = wa, 32..63 = wb
__global__ void pack_ew1(const float* __restrict__ ew1, _Float16* __restrict__ dst) {
    int i = blockIdx.x * blockDim.x + threadIdx.x;
    if (i >= Cd * 64) return;
    int k = i >> 6, j = i & 63;
    float v = (j < 32) ? ew1[k * 32 + j] : ew1[(Cd + k) * 32 + (j - 32)];
    dst[i] = (_Float16)v;
}

// ---------------------------------------------------------------- WMMA GEMM
// C[z][M x Ncols] = epilogue(A[z][M x K] @ B[z][K x Ncols]), row-major.
// blockDim = (32,4): 4 waves; wave w owns M-tile (blockIdx.y*4+w), block owns a
// 64-wide N stripe (blockIdx.x). Each wave: 4 accumulators, 4 WMMAs per A-frag.
// Requires: M % 64 == 0, Ncols % 64 == 0, K % 32 == 0, ldb % 16 == 0 (all call
// sites satisfy this), so there is no divergent exit around barriers and EXEC
// is all-ones for every WMMA.
__global__ void wmma_gemm(const _Float16* __restrict__ A, const _Float16* __restrict__ Bm,
                          const float* __restrict__ bias, const float* __restrict__ maskp,
                          void* __restrict__ Cout,
                          int M, int Ncols, int K, int lda, int ldb, int ldc,
                          long sA, long sB, long sC, int col_off, int flags) {
    // Double-buffered B tile, n-major: bs[buf][n*32 + k], 2 x 4 KB.
    __shared__ alignas(32) _Float16 bs[2][64 * 32];
    const int lane = threadIdx.x;                   // 0..31
    const int wave = threadIdx.y;                   // 0..3
    const int tid  = wave * 32 + lane;              // 0..127
    const int tn   = blockIdx.x;                    // 64-wide N stripe
    const int tm   = blockIdx.y * 4 + wave;         // 16-high M tile
    const int z    = blockIdx.z;

    const _Float16* Ab = A  + (size_t)z * sA;
    const _Float16* Bb = Bm + (size_t)z * sB;

    const int mrow = tm * 16 + (lane & 15);
    const int kh   = lane >> 4;       // half-wave id
    const int n    = lane & 15;

    // cooperative B-tile loader: 128 threads cover 32(K) x 64(N); one v16h each
    const int lk = tid >> 2;          // K row 0..31
    const int ln = (tid & 3) * 16;    // N base 0,16,32,48

    const int steps = K / 32;

    // preload tile 0 into buffer 0
    {
        v16h bv = *(const v16h*)(Bb + (size_t)lk * ldb + tn * 64 + ln);
#pragma unroll
        for (int i = 0; i < 16; ++i) bs[0][(ln + i) * 32 + lk] = bv[i];
    }

    v8f acc0 = {}, acc1 = {}, acc2 = {}, acc3 = {};
    int buf = 0;
    for (int kt = 0; kt < steps; ++kt) {
        __syncthreads();   // tile kt visible; reads of the other buffer retired

        const bool more = (kt + 1 < steps);
        v16h bnext = {};
        if (more)
            bnext = *(const v16h*)(Bb + (size_t)((kt + 1) * 32 + lk) * ldb + tn * 64 + ln);

        // A fragment: lane group kh holds K=kh*16+e (matches B)
        const int k0 = kt * 32;
        v16h af = *(const v16h*)(Ab + (size_t)mrow * lda + k0 + kh * 16);

        // B fragments (contiguous k order, same as A): lane group kh holds K=kh*16+e, col n
        const _Float16* bb = &bs[buf][0];
        v16h b0 = *(const v16h*)(bb + (0 * 16 + n) * 32 + kh * 16);
        v16h b1 = *(const v16h*)(bb + (1 * 16 + n) * 32 + kh * 16);
        v16h b2 = *(const v16h*)(bb + (2 * 16 + n) * 32 + kh * 16);
        v16h b3 = *(const v16h*)(bb + (3 * 16 + n) * 32 + kh * 16);

        acc0 = __builtin_amdgcn_wmma_f32_16x16x32_f16(false, af, false, b0, (short)0, acc0, false, false);
        acc1 = __builtin_amdgcn_wmma_f32_16x16x32_f16(false, af, false, b1, (short)0, acc1, false, false);
        acc2 = __builtin_amdgcn_wmma_f32_16x16x32_f16(false, af, false, b2, (short)0, acc2, false, false);
        acc3 = __builtin_amdgcn_wmma_f32_16x16x32_f16(false, af, false, b3, (short)0, acc3, false, false);

        if (more) {
            buf ^= 1;   // stage tile kt+1 into the buffer whose reads finished at the sync above
#pragma unroll
            for (int i = 0; i < 16; ++i) bs[buf][(ln + i) * 32 + lk] = bnext[i];
        }
    }

    // C/D layout: VGPR r, lanes 0-15 -> M=r, lanes 16-31 -> M=8+r; N=lane&15
    const int mb = tm * 16 + kh * 8;
    v8f accs[4] = {acc0, acc1, acc2, acc3};
#pragma unroll
    for (int t = 0; t < 4; ++t) {
        const int nc = tn * 64 + t * 16 + n;
        float bi = (flags & FLAG_BIAS) ? bias[nc] : 0.0f;
#pragma unroll
        for (int r = 0; r < 8; ++r) {
            float v = accs[t][r] + bi;
            int row = mb + r;
            if (flags & FLAG_MASK) v *= maskp[(size_t)z * M + row];
            if (flags & FLAG_RELU) v = fmaxf(v, 0.0f);
            size_t off = (size_t)z * sC + (size_t)row * ldc + col_off + nc;
            if (flags & FLAG_F16OUT) ((_Float16*)Cout)[off] = (_Float16)v;
            else                     ((float*)Cout)[off]    = v;
        }
    }
}

// ---------------------------------------------------------------- edge scores -> A_pred
// xab[b,n,0:32]=xa, [32:64]=xb. One thread per (b,i,j) with j>=i; computes the
// symmetric exp score and writes both (i,j) and (j,i). Diagonal = 0.
__global__ void edge_adj(const float* __restrict__ xab, const float* __restrict__ eb1,
                         const float* __restrict__ ew2, const float* __restrict__ eb2,
                         const float* __restrict__ mask, float* __restrict__ Apred) {
    long idx = (long)blockIdx.x * blockDim.x + threadIdx.x;
    if (idx >= (long)Bd * Nd * Nd) return;
    int j = (int)(idx % Nd);
    long t = idx / Nd;
    int i = (int)(t % Nd);
    int b = (int)(t / Nd);
    if (j < i) return;
    size_t base = (size_t)b * Nd;
    if (i == j) { Apred[(base + i) * Nd + j] = 0.0f; return; }
    const float* ra = xab + (base + i) * 64;
    const float* rb = xab + (base + j) * 64;
    float e0 = eb2[0], e1 = eb2[0];
#pragma unroll
    for (int k = 0; k < 32; ++k) {
        float w = ew2[k], c = eb1[k];
        float hij = ra[k] + rb[32 + k] + c;   // xa_i + xb_j + eb1
        float hji = rb[k] + ra[32 + k] + c;   // xa_j + xb_i + eb1
        e0 += fmaxf(hij, 0.0f) * w;
        e1 += fmaxf(hji, 0.0f) * w;
    }
    float y = __expf(0.5f * (e0 + e1));
    float v = (mask[base + i] > 0.0f && mask[base + j] > 0.0f) ? y : 0.0f;
    Apred[(base + i) * Nd + j] = v;
    Apred[(base + j) * Nd + i] = v;
}

// ---------------------------------------------------------------- Laplacian
// D[b,j] = rsqrt(colsum(M)[b,j] + 1 (identity) + 1e-5)   (axis=1 sum = column sum)
__global__ void lap_deg(const float* __restrict__ Mm, float* __restrict__ D) {
    int j = blockIdx.x * blockDim.x + threadIdx.x;
    int b = blockIdx.y;
    if (j >= Nd) return;
    const float* col = Mm + (size_t)b * Nd * Nd + j;
    float s = 0.0f;
    for (int i = 0; i < Nd; ++i) s += col[(size_t)i * Nd];
    D[b * Nd + j] = rsqrtf(s + 1.0f + 1e-5f);
}

// L[b,i,j] = D_i * (M + I)_ij * D_j, written directly as f16 GEMM operand
__global__ void lap_scale(const float* __restrict__ Mm, const float* __restrict__ D,
                          _Float16* __restrict__ L) {
    long idx = (long)blockIdx.x * blockDim.x + threadIdx.x;
    if (idx >= (long)Bd * Nd * Nd) return;
    int j = (int)(idx % Nd);
    long t = idx / Nd;
    int i = (int)(t % Nd);
    int b = (int)(t / Nd);
    float v = Mm[idx] + (i == j ? 1.0f : 0.0f);
    L[idx] = (_Float16)(D[b * Nd + i] * v * D[b * Nd + j]);
}

// ---------------------------------------------------------------- maxpool + FC
__global__ void maxpool_fc(const _Float16* __restrict__ h, const float* __restrict__ fcw,
                           const float* __restrict__ fcb, float* __restrict__ out) {
    __shared__ float g[64];
    int b = blockIdx.x, f = threadIdx.x;        // 64 threads
    const _Float16* hb = h + (size_t)b * Nd * 64 + f;
    float m = -3.4e38f;
    for (int n = 0; n < Nd; ++n) m = fmaxf(m, (float)hb[(size_t)n * 64]);
    g[f] = m;
    __syncthreads();
    if (f < 2) {
        float s = fcb[f];
        for (int k = 0; k < 64; ++k) s += g[k] * fcw[k * 2 + f];
        out[b * 2 + f] = s;
    }
}

// ---------------------------------------------------------------- launcher
extern "C" void kernel_launch(void* const* d_in, const int* in_sizes, int n_in,
                              void* d_out, int out_size, void* d_ws, size_t ws_size,
                              hipStream_t stream) {
    const float* x    = (const float*)d_in[0];
    const float* A    = (const float*)d_in[1];
    const float* mask = (const float*)d_in[2];
    const float* ew1  = (const float*)d_in[3];
    const float* eb1  = (const float*)d_in[4];
    const float* ew2  = (const float*)d_in[5];
    const float* eb2  = (const float*)d_in[6];
    const float* gw[3] = {(const float*)d_in[7], (const float*)d_in[9],  (const float*)d_in[11]};
    const float* gb[3] = {(const float*)d_in[8], (const float*)d_in[10], (const float*)d_in[12]};
    const float* fcw  = (const float*)d_in[13];
    const float* fcb  = (const float*)d_in[14];
    float* out = (float*)d_out;

    char* ws = (char*)d_ws;
    size_t off = 0;
    auto carve = [&](size_t bytes) -> char* {
        size_t a = (off + 255) & ~(size_t)255;
        off = a + bytes;
        return ws + a;
    };
    _Float16* x16   = (_Float16*)carve((size_t)Bd * Nd * Cd * 2);   // x in f16 (also layer-0 h)
    _Float16* ewB16 = (_Float16*)carve((size_t)Cd * 64 * 2);        // packed [wa|wb]
    float*    xab   = (float*)   carve((size_t)Bd * Nd * 64 * 4);   // [xa|xb] f32
    float*    Apred = (float*)   carve((size_t)Bd * Nd * Nd * 4);
    _Float16* L0    = (_Float16*)carve((size_t)Bd * Nd * Nd * 2);
    _Float16* L1    = (_Float16*)carve((size_t)Bd * Nd * Nd * 2);
    float*    Dbuf  = (float*)   carve((size_t)Bd * Nd * 4);
    _Float16* W16   = (_Float16*)carve((size_t)2 * Cd * 64 * 2);
    _Float16* xh16  = (_Float16*)carve((size_t)Bd * Nd * 2 * Cd * 2); // concat buffer
    _Float16* hA    = (_Float16*)carve((size_t)Bd * Nd * 64 * 2);
    _Float16* hB    = (_Float16*)carve((size_t)Bd * Nd * 64 * 2);

    dim3 blk(256);
    dim3 gblk(32, 4);

    int nx = Bd * Nd * Cd;
    f32_to_f16<<<(nx + 255) / 256, blk, 0, stream>>>(x, x16, nx);
    pack_ew1<<<(Cd * 64 + 255) / 256, blk, 0, stream>>>(ew1, ewB16);

    // [xa|xb] = x @ [wa|wb] : M=4096, N=64, K=256
    wmma_gemm<<<dim3(1, (Bd * Nd) / 64, 1), gblk, 0, stream>>>(
        x16, ewB16, nullptr, nullptr, xab,
        Bd * Nd, 64, Cd, Cd, 64, 64, 0, 0, 0, 0, 0);

    long tot = (long)Bd * Nd * Nd;
    edge_adj<<<(int)((tot + 255) / 256), blk, 0, stream>>>(xab, eb1, ew2, eb2, mask, Apred);

    lap_deg  <<<dim3((Nd + 255) / 256, Bd), blk, 0, stream>>>(A, Dbuf);
    lap_scale<<<(int)((tot + 255) / 256), blk, 0, stream>>>(A, Dbuf, L0);
    lap_deg  <<<dim3((Nd + 255) / 256, Bd), blk, 0, stream>>>(Apred, Dbuf);
    lap_scale<<<(int)((tot + 255) / 256), blk, 0, stream>>>(Apred, Dbuf, L1);

    const int Fin[3]  = {Cd, 64, 64};
    _Float16* hin = x16;
    _Float16* houts[3] = {hA, hB, hA};
    for (int l = 0; l < 3; ++l) {
        int fi = Fin[l], fo = 64;
        int wn = 2 * fi * fo;
        f32_to_f16<<<(wn + 255) / 256, blk, 0, stream>>>(gw[l], W16, wn);
        // P0 = L0 @ h -> xh[:, :, 0:fi]
        wmma_gemm<<<dim3(fi / 64, Nd / 64, Bd), gblk, 0, stream>>>(
            L0, hin, nullptr, nullptr, xh16,
            Nd, fi, Nd, Nd, fi, 2 * fi,
            (long)Nd * Nd, (long)Nd * fi, (long)Nd * 2 * fi, 0, FLAG_F16OUT);
        // P1 = L1 @ h -> xh[:, :, fi:2fi]
        wmma_gemm<<<dim3(fi / 64, Nd / 64, Bd), gblk, 0, stream>>>(
            L1, hin, nullptr, nullptr, xh16,
            Nd, fi, Nd, Nd, fi, 2 * fi,
            (long)Nd * Nd, (long)Nd * fi, (long)Nd * 2 * fi, fi, FLAG_F16OUT);
        // h' = relu((xh @ W + b) * m)
        wmma_gemm<<<dim3(fo / 64, Nd / 64, Bd), gblk, 0, stream>>>(
            xh16, W16, gb[l], mask, houts[l],
            Nd, fo, 2 * fi, 2 * fi, fo, fo,
            (long)Nd * 2 * fi, 0, (long)Nd * fo, 0,
            FLAG_F16OUT | FLAG_RELU | FLAG_BIAS | FLAG_MASK);
        hin = houts[l];
    }

    maxpool_fc<<<Bd, 64, 0, stream>>>(hin, fcw, fcb, out);

    (void)in_sizes; (void)n_in; (void)out_size; (void)ws_size;
}
